// LayoutModel_18846316495083
// MI455X (gfx1250) — matrix operand order repaired and checked by reading, not compiled, not run on hardware
//
#include <hip/hip_runtime.h>
#include <hip/hip_bf16.h>

typedef _Float16 f16;
typedef _Float16 v16h __attribute__((ext_vector_type(16)));
typedef _Float16 v8h  __attribute__((ext_vector_type(8)));
typedef float    v8f  __attribute__((ext_vector_type(8)));

#define CCFG   16
#define NNODE  10000
#define NEDGE  40000
#define KIN    224      /* 216 padded to multiple of 32 */
#define DMODEL 256
#define NCONV  4

/* ---------------- weight transforms ---------------- */
// conv_wl/conv_wr: [4][256(k)][256(n)] f32 -> WtL/WtR: [4][256(n)][256(k)] f16
__global__ void k_weights_conv(const float* __restrict__ wl, const float* __restrict__ wr,
                               f16* __restrict__ WtL, f16* __restrict__ WtR) {
    int n = blockIdx.x, i = blockIdx.y, k = threadIdx.x;
    size_t dst = ((size_t)i * 256 + n) * 256 + k;
    size_t src = ((size_t)i * 256 + k) * 256 + n;
    WtL[dst] = (f16)wl[src];
    WtR[dst] = (f16)wr[src];
}

// lin_w: [216(k)][256(n)] f32 -> linWt: [256(n)][224(k)] f16, zero-padded K
__global__ void k_weights_lin(const float* __restrict__ lw, f16* __restrict__ Wt) {
    int n = blockIdx.x, k = threadIdx.x;
    Wt[(size_t)n * KIN + k] = (k < 216) ? (f16)lw[(size_t)k * 256 + n] : (f16)0.f;
}

/* ---------------- feature build: concat(feat,cfg_emb,op_emb) -> f16 [C,N,224] ------- */
__global__ void k_feat(const int* __restrict__ cfg, const float* __restrict__ feat,
                       const int* __restrict__ op, const float* __restrict__ emb_op,
                       const float* __restrict__ emb_lay, f16* __restrict__ xin) {
    int n = blockIdx.x, c = blockIdx.y, t = threadIdx.x;
    float v;
    if (t < 140) {
        v = feat[(size_t)n * 140 + t];
    } else if (t < 212) {
        int u = t - 140, j = u >> 2, d = u & 3;
        int e = cfg[((size_t)c * NNODE + n) * 18 + j];
        v = emb_lay[e * 4 + d];
    } else if (t < 216) {
        v = emb_op[op[n] * 4 + (t - 212)];
    } else {
        v = 0.f;
    }
    xin[((size_t)c * NNODE + n) * KIN + t] = (f16)v;
}

/* ---------------- CSR build ---------------- */
__global__ void k_zero_i2(int* __restrict__ a, int* __restrict__ b, int n) {
    int i = blockIdx.x * blockDim.x + threadIdx.x;
    if (i < n) { a[i] = 0; b[i] = 0; }
}
__global__ void k_count(const int* __restrict__ ei, int* __restrict__ deg) {
    int e = blockIdx.x * blockDim.x + threadIdx.x;
    if (e < NEDGE) atomicAdd(&deg[ei[NEDGE + e]], 1);
}
__global__ void k_scan(const int* __restrict__ deg, int* __restrict__ rowstart,
                       float* __restrict__ invdeg) {
    // single-thread serial scan: 10k iters, negligible vs GEMMs
    int acc = 0;
    rowstart[0] = 0;
    for (int n = 0; n < NNODE; n++) {
        acc += deg[n];
        rowstart[n + 1] = acc;
        invdeg[n] = 1.f / fmaxf((float)deg[n], 1.f);
    }
}
__global__ void k_fill(const int* __restrict__ ei, const int* __restrict__ rowstart,
                       int* __restrict__ cursor, int* __restrict__ col) {
    int e = blockIdx.x * blockDim.x + threadIdx.x;
    if (e < NEDGE) {
        int d = ei[NEDGE + e], s = ei[e];
        int p = atomicAdd(&cursor[d], 1);
        col[rowstart[d] + p] = s;
    }
}

/* ---------------- neighbor gather-mean (deterministic set, f32 accum) -------------- */
__global__ __launch_bounds__(256) void k_agg(const f16* __restrict__ x,
                                             const int* __restrict__ rowstart,
                                             const int* __restrict__ col,
                                             const float* __restrict__ invdeg,
                                             f16* __restrict__ agg) {
    int n = blockIdx.x, c = blockIdx.y, t = threadIdx.x;
    int b = rowstart[n], e2 = rowstart[n + 1];
    float s = 0.f;
    for (int e = b; e < e2; e++) {
        int src = col[e];
        s += (float)x[((size_t)c * NNODE + src) * DMODEL + t];
    }
    agg[((size_t)c * NNODE + n) * DMODEL + t] = (f16)(s * invdeg[n]);
}

/* ---------------- WMMA GEMM: Out = act(A1@W1 + [A2@W2] + bias), f16 in / f32 acc --- */
__device__ inline v16h load_a_frag(const f16* p) {
    v8h lo = *(const v8h*)p;
    v8h hi = *(const v8h*)(p + 8);
    return __builtin_shufflevector(lo, hi, 0,1,2,3,4,5,6,7,8,9,10,11,12,13,14,15);
}

__global__ __launch_bounds__(256) void k_gemm(
        const f16* __restrict__ A1, const f16* __restrict__ W1, int K1,
        const f16* __restrict__ A2, const f16* __restrict__ W2, int K2,
        const float* __restrict__ bias, f16* __restrict__ Out, int do_relu) {
    int tid = threadIdx.x;
    int lane = tid & 31;
    int wid = tid >> 5;
    int lm = lane & 15, lh = lane >> 4;
    size_t m0 = (size_t)blockIdx.x * 64 + (size_t)(wid & 3) * 16;
    int n0 = blockIdx.y * 64 + (wid >> 2) * 32;

    float b0s = bias[n0 + lm], b1s = bias[n0 + 16 + lm];
    v8f acc0, acc1;
#pragma unroll
    for (int r = 0; r < 8; r++) { acc0[r] = b0s; acc1[r] = b1s; }

    {
        const f16* arow = A1 + (m0 + lm) * (size_t)K1 + lh * 16;
        const f16* w0p  = W1 + (size_t)(n0 + lm) * K1 + lh * 16;
        const f16* w1p  = W1 + (size_t)(n0 + 16 + lm) * K1 + lh * 16;
        for (int k = 0; k < K1; k += 32) {
            v16h a  = load_a_frag(arow + k);
            v16h b0 = *(const v16h*)(w0p + k);
            v16h b1 = *(const v16h*)(w1p + k);
            acc0 = __builtin_amdgcn_wmma_f32_16x16x32_f16(false, a, false, b0, (short)0, acc0, false, false);
            acc1 = __builtin_amdgcn_wmma_f32_16x16x32_f16(false, a, false, b1, (short)0, acc1, false, false);
        }
    }
    if (A2) {
        const f16* arow = A2 + (m0 + lm) * (size_t)K2 + lh * 16;
        const f16* w0p  = W2 + (size_t)(n0 + lm) * K2 + lh * 16;
        const f16* w1p  = W2 + (size_t)(n0 + 16 + lm) * K2 + lh * 16;
        for (int k = 0; k < K2; k += 32) {
            v16h a  = load_a_frag(arow + k);
            v16h b0 = *(const v16h*)(w0p + k);
            v16h b1 = *(const v16h*)(w1p + k);
            acc0 = __builtin_amdgcn_wmma_f32_16x16x32_f16(false, a, false, b0, (short)0, acc0, false, false);
            acc1 = __builtin_amdgcn_wmma_f32_16x16x32_f16(false, a, false, b1, (short)0, acc1, false, false);
        }
    }

#pragma unroll
    for (int r = 0; r < 8; r++) {
        float v0 = acc0[r], v1 = acc1[r];
        if (do_relu) { v0 = fmaxf(v0, 0.f); v1 = fmaxf(v1, 0.f); }
        size_t row = m0 + r + lh * 8;
        Out[row * DMODEL + n0 + lm]      = (f16)v0;
        Out[row * DMODEL + n0 + 16 + lm] = (f16)v1;
    }
}

/* ---------------- pooling + MLP head (f32 exact) ---------------- */
__global__ void k_zero_f(float* __restrict__ p, int n) {
    int i = blockIdx.x * blockDim.x + threadIdx.x;
    if (i < n) p[i] = 0.f;
}
__global__ __launch_bounds__(256) void k_pool(const f16* __restrict__ x, float* __restrict__ g) {
    int chunk = blockIdx.x, c = blockIdx.y, t = threadIdx.x;
    int n0 = chunk * 250;
    float s = 0.f;
    for (int n = n0; n < n0 + 250; n++)
        s += (float)x[((size_t)c * NNODE + n) * DMODEL + t];
    atomicAdd(&g[c * DMODEL + t], s);
}
__global__ __launch_bounds__(256) void k_mlp(const float* __restrict__ in, const float* __restrict__ w,
                                             const float* __restrict__ b, float* __restrict__ out,
                                             float scale) {
    int c = blockIdx.x, t = threadIdx.x;
    float s = b[t];
    for (int k = 0; k < DMODEL; k++)
        s += (in[c * DMODEL + k] * scale) * w[(size_t)k * DMODEL + t];
    out[c * DMODEL + t] = fmaxf(s, 0.f);
}
__global__ void k_mlp3(const float* __restrict__ h2, const float* __restrict__ w3,
                       const float* __restrict__ b3, float* __restrict__ out) {
    int c = threadIdx.x;
    if (c < CCFG) {
        float s = b3[0];
        for (int k = 0; k < DMODEL; k++) s += h2[c * DMODEL + k] * w3[k];
        out[c] = s;
    }
}

/* ---------------- launch ---------------- */
extern "C" void kernel_launch(void* const* d_in, const int* in_sizes, int n_in,
                              void* d_out, int out_size, void* d_ws, size_t ws_size,
                              hipStream_t stream) {
    (void)in_sizes; (void)n_in; (void)out_size; (void)ws_size;

    const int*   x_node_cfg = (const int*)  d_in[0];
    const float* x_feat     = (const float*)d_in[1];
    const int*   x_op       = (const int*)  d_in[2];
    const int*   edge_index = (const int*)  d_in[3];
    const float* emb_op     = (const float*)d_in[4];
    const float* emb_layout = (const float*)d_in[5];
    const float* lin_w      = (const float*)d_in[6];
    const float* lin_b      = (const float*)d_in[7];
    const float* conv_wl    = (const float*)d_in[8];
    const float* conv_bl    = (const float*)d_in[9];
    const float* conv_wr    = (const float*)d_in[10];
    const float* w1         = (const float*)d_in[11];
    const float* b1         = (const float*)d_in[12];
    const float* w2         = (const float*)d_in[13];
    const float* b2         = (const float*)d_in[14];
    const float* w3         = (const float*)d_in[15];
    const float* b3         = (const float*)d_in[16];
    float* out = (float*)d_out;

    // bump allocator over d_ws (256B aligned chunks)
    char* base = (char*)d_ws;
    size_t off = 0;
    auto alloc = [&](size_t bytes) -> void* {
        void* p = base + off;
        off += (bytes + 255) & ~(size_t)255;
        return p;
    };
    const size_t MROWS = (size_t)CCFG * NNODE;           // 160000 (divisible by 64)
    f16*   xin   = (f16*)  alloc(MROWS * KIN * sizeof(f16));
    f16*   xA    = (f16*)  alloc(MROWS * DMODEL * sizeof(f16));
    f16*   xB    = (f16*)  alloc(MROWS * DMODEL * sizeof(f16));
    f16*   agg   = (f16*)  alloc(MROWS * DMODEL * sizeof(f16));
    f16*   WtL   = (f16*)  alloc((size_t)NCONV * 256 * 256 * sizeof(f16));
    f16*   WtR   = (f16*)  alloc((size_t)NCONV * 256 * 256 * sizeof(f16));
    f16*   linWt = (f16*)  alloc((size_t)256 * KIN * sizeof(f16));
    int*   deg      = (int*)  alloc(NNODE * sizeof(int));
    int*   cursor   = (int*)  alloc(NNODE * sizeof(int));
    int*   rowstart = (int*)  alloc((NNODE + 1) * sizeof(int));
    int*   col      = (int*)  alloc(NEDGE * sizeof(int));
    float* invdeg   = (float*)alloc(NNODE * sizeof(float));
    float* g        = (float*)alloc(CCFG * DMODEL * sizeof(float));
    float* h1       = (float*)alloc(CCFG * DMODEL * sizeof(float));
    float* h2       = (float*)alloc(CCFG * DMODEL * sizeof(float));

    // 1. weight transforms (f32 -> transposed f16)
    k_weights_conv<<<dim3(256, NCONV), 256, 0, stream>>>(conv_wl, conv_wr, WtL, WtR);
    k_weights_lin <<<dim3(256), KIN, 0, stream>>>(lin_w, linWt);

    // 2. feature build
    k_feat<<<dim3(NNODE, CCFG), KIN, 0, stream>>>(x_node_cfg, x_feat, x_op, emb_op, emb_layout, xin);

    // 3. CSR build
    k_zero_i2<<<(NNODE + 255) / 256, 256, 0, stream>>>(deg, cursor, NNODE);
    k_count<<<(NEDGE + 255) / 256, 256, 0, stream>>>(edge_index, deg);
    k_scan<<<1, 1, 0, stream>>>(deg, rowstart, invdeg);
    k_fill<<<(NEDGE + 255) / 256, 256, 0, stream>>>(edge_index, rowstart, cursor, col);

    // 4. input linear: xA = xin @ lin_w + lin_b   (no activation)
    dim3 ggrid((unsigned)(MROWS / 64), DMODEL / 64);
    k_gemm<<<ggrid, 256, 0, stream>>>(xin, linWt, KIN, nullptr, nullptr, 0, lin_b, xA, 0);

    // 5. SAGE conv layers: x = relu(agg @ Wl + bl + x @ Wr)
    f16* cur = xA;
    f16* nxt = xB;
    for (int i = 0; i < NCONV; i++) {
        k_agg<<<dim3(NNODE, CCFG), 256, 0, stream>>>(cur, rowstart, col, invdeg, agg);
        k_gemm<<<ggrid, 256, 0, stream>>>(agg, WtL + (size_t)i * 256 * 256, DMODEL,
                                          cur, WtR + (size_t)i * 256 * 256, DMODEL,
                                          conv_bl + (size_t)i * DMODEL, nxt, 1);
        f16* t = cur; cur = nxt; nxt = t;
    }

    // 6. mean pool + MLP head
    k_zero_f<<<(CCFG * DMODEL + 255) / 256, 256, 0, stream>>>(g, CCFG * DMODEL);
    k_pool<<<dim3(NNODE / 250, CCFG), 256, 0, stream>>>(cur, g);
    k_mlp <<<CCFG, 256, 0, stream>>>(g,  w1, b1, h1, 1.f / (float)NNODE);
    k_mlp <<<CCFG, 256, 0, stream>>>(h1, w2, b2, h2, 1.f);
    k_mlp3<<<1, 32, 0, stream>>>(h2, w3, b3, out);
}
